// EncoderDecoder_9895604650575
// MI455X (gfx1250) — matrix operand order repaired and checked by reading, not compiled, not run on hardware
//
#include <hip/hip_runtime.h>
#include <hip/hip_bf16.h>
#include <math.h>
#include <stdint.h>

// ---------------------------------------------------------------------------
// Types for CDNA5 WMMA
// ---------------------------------------------------------------------------
typedef __attribute__((ext_vector_type(16))) _Float16 v16h;
typedef __attribute__((ext_vector_type(8)))  float    v8f;

#define D_MODEL 256
#define FF_DIM  1024
#define N_HEADS 8
#define HEAD_DIM 32
#define SEQ     4096
#define BATCH   2
#define M_ROWS  (BATCH * SEQ)       // 8192
#define NBH     (BATCH * N_HEADS)   // 16
#define NCHUNK  (SEQ / 32)          // 128 causal chunks
#define NSLICE  8                   // S-slices for non-causal KV reduction
#define ATTN_EPS 1e-6f
#define LN_EPS   1e-5f

#define ACT_NONE 0
#define ACT_RELU 1
#define ACT_FM   2   // elu(x)+1  ==  x>0 ? x+1 : exp(x)

// ---------------------------------------------------------------------------
// f32 -> f16 elementwise conversion (weights once per launch; n % 4 == 0)
// ---------------------------------------------------------------------------
__global__ __launch_bounds__(256)
void cvt_f32_to_f16_kernel(const float* __restrict__ in, _Float16* __restrict__ out, int n)
{
    int i = (blockIdx.x * 256 + threadIdx.x) * 4;
    if (i >= n) return;
    #pragma unroll
    for (int e = 0; e < 4; ++e) out[i + e] = (_Float16)in[i + e];
}

// ---------------------------------------------------------------------------
// WMMA GEMM (f16 operands, f32 accumulate), double-buffered ASYNC staging:
//   C[M,N] = act( A[M,K] @ W[K,N] + bias[N] );  optional f16 copy Ch.
// 256 threads = 8 waves; block tile 128(M) x 64(N); K-step 32.
// Tiles move global->LDS with GLOBAL_LOAD_ASYNC_TO_LDS_B128 (ASYNCcnt),
// 3 async instrs per wave per stage; s_wait_asynccnt 3 overlaps stages.
// ---------------------------------------------------------------------------
#define BM 128
#define BN 64
#define BK 32
#define LDA (BK + 8)    // f16 elems; row stride 80B (16B multiple)
#define LDB (BN + 8)    // f16 elems; row stride 144B (16B multiple)

__device__ __forceinline__ void async_b128(uint32_t lds_off, const void* g)
{
    asm volatile("global_load_async_to_lds_b128 %0, %1, off"
                 :: "v"(lds_off), "v"((unsigned long long)(uintptr_t)g)
                 : "memory");
}

__global__ __launch_bounds__(256)
void gemm_wmma_f16_kernel(const _Float16* __restrict__ Ah, const _Float16* __restrict__ Wh,
                          const float* __restrict__ bias, float* __restrict__ C,
                          _Float16* __restrict__ Ch, int M, int K, int N, int act)
{
    __shared__ __align__(16) _Float16 sA[2][BM][LDA];
    __shared__ __align__(16) _Float16 sB[2][BK][LDB];

    const int tid  = threadIdx.x;
    const int lane = tid & 31;
    const int wid  = tid >> 5;
    const int wm   = (wid & 3) * 32;
    const int wn   = (wid >> 2) * 32;

    const int m0  = blockIdx.x * BM;
    const int n0c = blockIdx.y * BN;

    // async copy lane mapping: A tile = 512 x 16B segs (2 instrs/wave),
    // W tile = 256 x 16B segs (1 instr/wave)
    const int aseg0 = wid * 64 + lane;
    const int wseg  = wid * 32 + lane;
    const int wrow  = wseg >> 3, wcs = wseg & 7;

    const int arow = lane & 15;
    const int kb   = (lane >> 4) * 8;

    auto issue = [&](int buf, int k0) {
        #pragma unroll
        for (int q = 0; q < 2; ++q) {
            const int seg = aseg0 + q * 32;
            const int r = seg >> 2, cs = seg & 3;
            async_b128((uint32_t)(uintptr_t)&sA[buf][r][cs * 8],
                       Ah + (size_t)(m0 + r) * K + k0 + cs * 8);
        }
        async_b128((uint32_t)(uintptr_t)&sB[buf][wrow][wcs * 8],
                   Wh + (size_t)(k0 + wrow) * N + n0c + wcs * 8);
    };

    v8f acc[2][2];
    acc[0][0] = {}; acc[0][1] = {}; acc[1][0] = {}; acc[1][1] = {};

    const int KT = K / BK;
    issue(0, 0);
    for (int kt = 0; kt < KT; ++kt) {
        const int buf = kt & 1;
        if (kt + 1 < KT) {
            issue(buf ^ 1, (kt + 1) * BK);
            // 3 just-issued may stay outstanding; in-order completion ->
            // previous stage's copies are done.
            asm volatile("s_wait_asynccnt 0x3" ::: "memory");
        } else {
            asm volatile("s_wait_asynccnt 0x0" ::: "memory");
        }
        __syncthreads();

        v16h af[2], bf[2];
        #pragma unroll
        for (int mt = 0; mt < 2; ++mt) {
            const _Float16* p0 = &sA[buf][wm + mt * 16 + arow][kb];
            const _Float16* p1 = &sA[buf][wm + mt * 16 + arow][16 + kb];
            #pragma unroll
            for (int e = 0; e < 8; ++e) { af[mt][e] = p0[e]; af[mt][8 + e] = p1[e]; }
        }
        #pragma unroll
        for (int nt = 0; nt < 2; ++nt) {
            const _Float16* p = &sB[buf][kb][wn + nt * 16 + arow];
            #pragma unroll
            for (int e = 0; e < 16; ++e) bf[nt][e] = p[((e & 7) + (e >> 3) * 16) * LDB];
        }

        #pragma unroll
        for (int mt = 0; mt < 2; ++mt)
            #pragma unroll
            for (int nt = 0; nt < 2; ++nt)
                acc[mt][nt] = __builtin_amdgcn_wmma_f32_16x16x32_f16(
                    false, af[mt], false, bf[nt], (short)0, acc[mt][nt],
                    false, false);
        __syncthreads();
    }

    #pragma unroll
    for (int mt = 0; mt < 2; ++mt) {
        #pragma unroll
        for (int nt = 0; nt < 2; ++nt) {
            const int n  = n0c + wn + nt * 16 + (lane & 15);
            const float bv = bias ? bias[n] : 0.0f;
            #pragma unroll
            for (int r = 0; r < 8; ++r) {
                const int m = m0 + wm + mt * 16 + r + (lane >> 4) * 8;
                float v = acc[mt][nt][r] + bv;
                if (act == ACT_RELU)      v = fmaxf(v, 0.0f);
                else if (act == ACT_FM)   v = (v > 0.0f) ? (v + 1.0f) : __expf(v);
                C[(size_t)m * N + n] = v;
                if (Ch) Ch[(size_t)m * N + n] = (_Float16)v;
            }
        }
    }
}

// ---------------------------------------------------------------------------
// Embedding: out[M,256] = X[M,F] @ W[F,256] + b  (dual f32 + f16 output)
// ---------------------------------------------------------------------------
__global__ __launch_bounds__(256)
void embed_kernel(const float* __restrict__ X, const float* __restrict__ W,
                  const float* __restrict__ b, float* __restrict__ out,
                  _Float16* __restrict__ outh, int F, int M)
{
    int idx = blockIdx.x * 256 + threadIdx.x;
    if (idx >= M * D_MODEL) return;
    int row = idx >> 8, col = idx & 255;
    float a = b[col];
    for (int f = 0; f < F; ++f) a += X[(size_t)row * F + f] * W[f * D_MODEL + col];
    out[idx]  = a;
    outh[idx] = (_Float16)a;
}

// ---------------------------------------------------------------------------
// Final projection: out[M,3] = X[M,256] @ W[256,3] + b
// ---------------------------------------------------------------------------
__global__ __launch_bounds__(256)
void final_kernel(const float* __restrict__ X, const float* __restrict__ W,
                  const float* __restrict__ b, float* __restrict__ out, int M)
{
    int idx = blockIdx.x * 256 + threadIdx.x;
    if (idx >= M * 3) return;
    int row = idx / 3, c = idx % 3;
    float a = b[c];
    const float* xp = X + (size_t)row * D_MODEL;
    for (int k = 0; k < D_MODEL; ++k) a += xp[k] * W[k * 3 + c];
    out[idx] = a;
}

// ---------------------------------------------------------------------------
// LayerNorm(+residual): out = LN(x [+ y]) * g + b.  One wave per row.
// Dual f32 + f16 output (f16 feeds downstream GEMM A-operands).
// ---------------------------------------------------------------------------
__global__ __launch_bounds__(256)
void ln_residual_kernel(const float* __restrict__ x, const float* __restrict__ y,
                        const float* __restrict__ g, const float* __restrict__ b,
                        float* __restrict__ out, _Float16* __restrict__ outh, int M)
{
    int wave = threadIdx.x >> 5, lane = threadIdx.x & 31;
    int row  = blockIdx.x * 8 + wave;
    if (row >= M) return;
    const size_t base = (size_t)row * D_MODEL;

    float v[8], s = 0.0f;
    #pragma unroll
    for (int i = 0; i < 8; ++i) {
        int c = lane + i * 32;
        v[i] = x[base + c] + (y ? y[base + c] : 0.0f);
        s += v[i];
    }
    #pragma unroll
    for (int off = 16; off; off >>= 1) s += __shfl_xor(s, off, 32);
    const float mu = s * (1.0f / D_MODEL);

    float vs = 0.0f;
    #pragma unroll
    for (int i = 0; i < 8; ++i) { float d = v[i] - mu; vs += d * d; }
    #pragma unroll
    for (int off = 16; off; off >>= 1) vs += __shfl_xor(vs, off, 32);
    const float rs = rsqrtf(vs * (1.0f / D_MODEL) + LN_EPS);

    #pragma unroll
    for (int i = 0; i < 8; ++i) {
        int c = lane + i * 32;
        float o = (v[i] - mu) * rs * g[c] + b[c];
        out[base + c]  = o;
        outh[base + c] = (_Float16)o;
    }
}

// ---------------------------------------------------------------------------
// Non-causal KV reduction, pass A (WMMA):
//   PartKV[bh][sc] = sum_{s in slice} K_s^T V_s ;  PartKS = sum K_s
// Grid (NBH, NSLICE); 8 waves = 4 output tiles x 2 S-halves; deterministic
// in-block LDS combine.
// ---------------------------------------------------------------------------
__global__ __launch_bounds__(256)
void linkv_part_kernel(const float* __restrict__ Kg, const float* __restrict__ Vg,
                       float* __restrict__ PartKV, float* __restrict__ PartKS, int S)
{
    __shared__ _Float16 sKT[32][72];
    __shared__ _Float16 sV [64][40];
    __shared__ float    red[2][32][33];

    const int bh = blockIdx.x, sc = blockIdx.y;
    const int bb = bh >> 3, h = bh & 7;
    const size_t base = ((size_t)bb * S) * D_MODEL + h * HEAD_DIM;

    const int t     = threadIdx.x;
    const int lane  = t & 31;
    const int w     = t >> 5;
    const int mt    = w & 1;
    const int nt    = (w >> 1) & 1;
    const int shalf = w >> 2;

    const int arow = lane & 15;
    const int kb   = (lane >> 4) * 8;

    const int ss = t >> 2;
    const int d0 = (t & 3) * 8;

    v8f acc = {};
    float ksacc = 0.0f;

    const int slice = S / NSLICE;
    for (int it = 0; it < slice / 64; ++it) {
        const int sb = sc * slice + it * 64;
        const float* kp = Kg + base + (size_t)(sb + ss) * D_MODEL + d0;
        const float* vp = Vg + base + (size_t)(sb + ss) * D_MODEL + d0;
        #pragma unroll
        for (int e = 0; e < 8; ++e) sKT[d0 + e][ss] = (_Float16)kp[e];
        #pragma unroll
        for (int e = 0; e < 8; ++e) sV[ss][d0 + e]  = (_Float16)vp[e];
        __syncthreads();

        if (t < 32) {
            #pragma unroll
            for (int s = 0; s < 64; ++s) ksacc += (float)sKT[t][s];
        }

        v16h af, bf;
        {
            const _Float16* p0 = &sKT[mt * 16 + arow][shalf * 32 + kb];
            const _Float16* p1 = &sKT[mt * 16 + arow][shalf * 32 + 16 + kb];
            #pragma unroll
            for (int e = 0; e < 8; ++e) { af[e] = p0[e]; af[8 + e] = p1[e]; }
        }
        {
            const _Float16* p = &sV[shalf * 32 + kb][nt * 16 + arow];
            #pragma unroll
            for (int e = 0; e < 16; ++e) bf[e] = p[((e & 7) + (e >> 3) * 16) * 40];
        }
        acc = __builtin_amdgcn_wmma_f32_16x16x32_f16(false, af, false, bf,
                                                     (short)0, acc, false, false);
        __syncthreads();
    }

    #pragma unroll
    for (int r = 0; r < 8; ++r)
        red[shalf][mt * 16 + r + (lane >> 4) * 8][nt * 16 + (lane & 15)] = acc[r];
    __syncthreads();

    {
        const int idx4 = t * 4;
        const int d = idx4 >> 5, m = idx4 & 31;
        float* o = PartKV + ((size_t)bh * NSLICE + sc) * 1024 + d * 32 + m;
        #pragma unroll
        for (int e = 0; e < 4; ++e) o[e] = red[0][d][m + e] + red[1][d][m + e];
    }
    if (t < 32) PartKS[((size_t)bh * NSLICE + sc) * 32 + t] = ksacc;
}

__global__ __launch_bounds__(256)
void linkv_reduce_kernel(const float* __restrict__ PartKV, const float* __restrict__ PartKS,
                         float* __restrict__ KV, float* __restrict__ KS)
{
    int idx = blockIdx.x * 256 + threadIdx.x;
    if (idx >= NBH * 1056) return;
    int bh = idx / 1056, r = idx % 1056;
    if (r < 1024) {
        float a = 0.0f;
        for (int c = 0; c < NSLICE; ++c)
            a += PartKV[((size_t)bh * NSLICE + c) * 1024 + r];
        KV[(size_t)bh * 1024 + r] = a;
    } else {
        int d = r - 1024;
        float a = 0.0f;
        for (int c = 0; c < NSLICE; ++c)
            a += PartKS[((size_t)bh * NSLICE + c) * 32 + d];
        KS[bh * 32 + d] = a;
    }
}

// ---------------------------------------------------------------------------
// Non-causal apply: one wave per (token, head). Dual f32 + f16 output.
// ---------------------------------------------------------------------------
__global__ __launch_bounds__(256)
void lin_apply_kernel(const float* __restrict__ Qg, const float* __restrict__ KV,
                      const float* __restrict__ Ksum, float* __restrict__ Og,
                      _Float16* __restrict__ Oh, int S)
{
    const int unit = blockIdx.x * 8 + (threadIdx.x >> 5);
    const int lane = threadIdx.x & 31;
    const int h  = unit & 7;
    const int bl = unit >> 3;
    const int bh = (bl / S) * 8 + h;

    const size_t base = (size_t)bl * D_MODEL + h * HEAD_DIM;
    const float q = Qg[base + lane];

    float zd = q * Ksum[bh * 32 + lane];
    #pragma unroll
    for (int off = 16; off; off >>= 1) zd += __shfl_xor(zd, off, 32);
    const float z = 1.0f / (zd + ATTN_EPS);

    const float* kvp = KV + (size_t)bh * 1024;
    float num = 0.0f;
    #pragma unroll
    for (int d = 0; d < 32; ++d)
        num += __shfl(q, d, 32) * kvp[d * 32 + lane];

    const float o = num * z;
    Og[base + lane] = o;
    Oh[base + lane] = (_Float16)o;
}

// ---------------------------------------------------------------------------
// Causal pass 1 (WMMA): ChunkKV[bh][c] = Kc^T @ Vc ; ChunkKS = colsum(Kc)
// ---------------------------------------------------------------------------
__global__ __launch_bounds__(256)
void causal_chunk_kv_kernel(const float* __restrict__ Kg, const float* __restrict__ Vg,
                            float* __restrict__ ChunkKV, float* __restrict__ ChunkKS, int S)
{
    __shared__ _Float16 sKT[32][40];
    __shared__ _Float16 sV [32][40];

    const int bh = blockIdx.x, c = blockIdx.y;
    const int bb = bh >> 3, h = bh & 7;
    const size_t base = ((size_t)bb * S) * D_MODEL + h * HEAD_DIM;

    const int t    = threadIdx.x;
    const int lane = t & 31;
    const int w    = t >> 5;

    {
        const int s = t >> 3, d0 = (t & 7) * 4;
        const float* kp = Kg + base + (size_t)(c * 32 + s) * D_MODEL + d0;
        const float* vp = Vg + base + (size_t)(c * 32 + s) * D_MODEL + d0;
        #pragma unroll
        for (int e = 0; e < 4; ++e) sKT[d0 + e][s] = (_Float16)kp[e];
        #pragma unroll
        for (int e = 0; e < 4; ++e) sV[s][d0 + e]  = (_Float16)vp[e];
    }
    __syncthreads();

    if (w < 4) {
        const int mt = w & 1, nt = w >> 1;
        const int arow = lane & 15, kb = (lane >> 4) * 8;
        v16h af, bf;
        {
            const _Float16* p0 = &sKT[mt * 16 + arow][kb];
            const _Float16* p1 = &sKT[mt * 16 + arow][16 + kb];
            #pragma unroll
            for (int e = 0; e < 8; ++e) { af[e] = p0[e]; af[8 + e] = p1[e]; }
        }
        {
            const _Float16* p = &sV[kb][nt * 16 + arow];
            #pragma unroll
            for (int e = 0; e < 16; ++e) bf[e] = p[((e & 7) + (e >> 3) * 16) * 40];
        }
        v8f acc = {};
        acc = __builtin_amdgcn_wmma_f32_16x16x32_f16(false, af, false, bf,
                                                     (short)0, acc, false, false);
        float* o = ChunkKV + ((size_t)bh * NCHUNK + c) * 1024;
        #pragma unroll
        for (int r = 0; r < 8; ++r) {
            const int d = mt * 16 + r + (lane >> 4) * 8;
            const int m = nt * 16 + (lane & 15);
            o[d * 32 + m] = acc[r];
        }
    } else if (w == 4) {
        float s2 = 0.0f;
        #pragma unroll
        for (int s = 0; s < 32; ++s) s2 += (float)sKT[lane][s];
        ChunkKS[((size_t)bh * NCHUNK + c) * 32 + lane] = s2;
    }
}

// ---------------------------------------------------------------------------
// Causal pass 2: in-place exclusive prefix sum over chunks per (b,h).
// ---------------------------------------------------------------------------
__global__ __launch_bounds__(256)
void causal_scan_kernel(float* __restrict__ ChunkKV, float* __restrict__ ChunkKS)
{
    int idx = blockIdx.x * 256 + threadIdx.x;
    if (idx >= NBH * 1056) return;
    int bh = idx / 1056, r = idx % 1056;
    if (r < 1024) {
        float* p = ChunkKV + (size_t)bh * NCHUNK * 1024 + r;
        float run = 0.0f;
        for (int c = 0; c < NCHUNK; ++c) {
            float tv = p[(size_t)c * 1024];
            p[(size_t)c * 1024] = run;
            run += tv;
        }
    } else {
        float* p = ChunkKS + (size_t)bh * NCHUNK * 32 + (r - 1024);
        float run = 0.0f;
        for (int c = 0; c < NCHUNK; ++c) {
            float tv = p[c * 32];
            p[c * 32] = run;
            run += tv;
        }
    }
}

// ---------------------------------------------------------------------------
// Causal pass 3: per-chunk apply. Dual f32 + f16 output; O may alias Qg.
// ---------------------------------------------------------------------------
__global__ __launch_bounds__(256)
void causal_apply_kernel(const float* __restrict__ Qg, const float* __restrict__ Kg,
                         const float* __restrict__ Vg,
                         const float* __restrict__ StateKV, const float* __restrict__ StateKS,
                         float* __restrict__ Og, _Float16* __restrict__ Oh, int S)
{
    __shared__ float sQ[32][33], sK[32][33], sV[32][33], sS[32][33], sKV[32][33];
    __shared__ float sks[32], denom[32];

    const int bh = blockIdx.x, c = blockIdx.y;
    const int bb = bh >> 3, h = bh & 7;
    const size_t base = ((size_t)bb * S) * D_MODEL + h * HEAD_DIM;

    const int t  = threadIdx.x;
    const int i  = t >> 3;
    const int j0 = (t & 7) * 4;

    const size_t rb = base + (size_t)(c * 32 + i) * D_MODEL + j0;
    const float* stkv = StateKV + ((size_t)bh * NCHUNK + c) * 1024;
    #pragma unroll
    for (int e = 0; e < 4; ++e) {
        sQ[i][j0 + e]  = Qg[rb + e];
        sK[i][j0 + e]  = Kg[rb + e];
        sV[i][j0 + e]  = Vg[rb + e];
        sKV[i][j0 + e] = stkv[i * 32 + j0 + e];
    }
    if (t < 32) sks[t] = StateKS[((size_t)bh * NCHUNK + c) * 32 + t];
    __syncthreads();

    #pragma unroll
    for (int e = 0; e < 4; ++e) {
        int j = j0 + e;
        float s = 0.0f;
        if (j <= i)
            for (int d = 0; d < 32; ++d) s += sQ[i][d] * sK[j][d];
        sS[i][j] = s;
    }
    __syncthreads();

    float o[4];
    #pragma unroll
    for (int e = 0; e < 4; ++e) {
        int m = j0 + e;
        float a = 0.0f;
        for (int d = 0; d < 32; ++d) a += sQ[i][d] * sKV[d][m];
        for (int j = 0; j < 32; ++j) a += sS[i][j] * sV[j][m];
        o[e] = a;
    }
    if (t < 32) {
        float ds = 0.0f;
        for (int d = 0; d < 32; ++d) ds += sQ[t][d] * sks[d];
        for (int j = 0; j <= t; ++j) ds += sS[t][j];
        denom[t] = ds;
    }
    __syncthreads();

    const float z = 1.0f / (denom[i] + ATTN_EPS);
    #pragma unroll
    for (int e = 0; e < 4; ++e) {
        const float ov = o[e] * z;
        Og[rb + e] = ov;
        Oh[rb + e] = (_Float16)ov;
    }
}

// ---------------------------------------------------------------------------
// Host orchestration
// ---------------------------------------------------------------------------
static inline void gemm(const _Float16* A, const _Float16* W, const float* bias,
                        float* C, _Float16* Ch, int M, int K, int N, int act,
                        hipStream_t s)
{
    dim3 grid(M / BM, N / BN), block(256);
    gemm_wmma_f16_kernel<<<grid, block, 0, s>>>(A, W, bias, C, Ch, M, K, N, act);
}

extern "C" void kernel_launch(void* const* d_in, const int* in_sizes, int n_in,
                              void* d_out, int out_size, void* d_ws, size_t ws_size,
                              hipStream_t stream)
{
    (void)in_sizes; (void)n_in; (void)out_size; (void)ws_size;
    auto IN = [&](int i) { return (const float*)d_in[i]; };

    const float* src  = IN(0);
    const float* tgt  = IN(1);
    const float* seW  = IN(2); const float* seB = IN(3);
    const float* teW  = IN(4); const float* teB = IN(5);
    const float* fW   = IN(6); const float* fB  = IN(7);
    // enc layer l: base 8 + 16*l  -> [qW qb kW kb vW vb oW ob ln1g ln1b ln2g ln2b f1W f1b f2W f2b]
    // dec layer l: base 72 + 26*l -> [self(8) cross(8) ln1(2) ln2(2) ln3(2) f1(2) f2(2)]

    const int M = M_ROWS;
    const size_t MD = (size_t)M * D_MODEL;
    const size_t MF = (size_t)M * FF_DIM;

    // ---- fp32 scratch ----
    float* ws = (float*)d_ws;
    float* XE = ws;      ws += MD;
    float* XD = ws;      ws += MD;
    float* Qb = ws;      ws += MD;
    float* Kb = ws;      ws += MD;
    float* Vb = ws;      ws += MD;
    float* H1 = ws;      ws += MF;
    float* KV = ws;      ws += (size_t)NBH * 1024;
    float* KS = ws;      ws += (size_t)NBH * 32;
    float* PartKV  = ws; ws += (size_t)NBH * NSLICE * 1024;
    float* PartKS  = ws; ws += (size_t)NBH * NSLICE * 32;
    float* ChunkKV = ws; ws += (size_t)NBH * NCHUNK * 1024;
    float* ChunkKS = ws; ws += (size_t)NBH * NCHUNK * 32;

    // ---- f16 scratch (16B-aligned: all float block sizes are x4 floats) ----
    _Float16* hp = (_Float16*)ws;
    _Float16* XEh  = hp; hp += MD;
    _Float16* XDh  = hp; hp += MD;
    _Float16* H1h  = hp; hp += MF;
    _Float16* ATTh = hp; hp += MD;

    const dim3 b256(256);

    // ---- convert all GEMM weights to f16 (once per launch) ----
    const _Float16* W16[176] = {};
    auto addW = [&](int idx, int n) {
        _Float16* dst = hp; hp += n;
        cvt_f32_to_f16_kernel<<<(n / 4 + 255) / 256, b256, 0, stream>>>(IN(idx), dst, n);
        W16[idx] = dst;
    };
    for (int l = 0; l < 4; ++l) {
        const int p = 8 + 16 * l;
        addW(p + 0, 65536); addW(p + 2, 65536); addW(p + 4, 65536); addW(p + 6, 65536);
        addW(p + 12, D_MODEL * FF_DIM); addW(p + 14, FF_DIM * D_MODEL);
    }
    for (int l = 0; l < 4; ++l) {
        const int p = 72 + 26 * l;
        addW(p + 0, 65536); addW(p + 2, 65536); addW(p + 4, 65536); addW(p + 6, 65536);
        addW(p + 8, 65536); addW(p + 10, 65536); addW(p + 12, 65536); addW(p + 14, 65536);
        addW(p + 22, D_MODEL * FF_DIM); addW(p + 24, FF_DIM * D_MODEL);
    }

    const int lnGrid    = M / 8;
    const int applyGrid = (BATCH * SEQ * N_HEADS) / 8;
    const int scanGrid  = (NBH * 1056 + 255) / 256;

    auto noncausal = [&](const float* Q, const float* K_, const float* V_,
                         float* O, _Float16* Oh) {
        linkv_part_kernel<<<dim3(NBH, NSLICE), b256, 0, stream>>>(K_, V_, PartKV, PartKS, SEQ);
        linkv_reduce_kernel<<<scanGrid, b256, 0, stream>>>(PartKV, PartKS, KV, KS);
        lin_apply_kernel<<<applyGrid, b256, 0, stream>>>(Q, KV, KS, O, Oh, SEQ);
    };

    // ===================== Encoder =====================
    embed_kernel<<<(M * D_MODEL) / 256, b256, 0, stream>>>(src, seW, seB, XE, XEh, 4, M);

    for (int l = 0; l < 4; ++l) {
        const int p = 8 + 16 * l;
        gemm(XEh, W16[p + 0], IN(p + 1), Qb, nullptr, M, D_MODEL, D_MODEL, ACT_FM, stream);
        gemm(XEh, W16[p + 2], IN(p + 3), Kb, nullptr, M, D_MODEL, D_MODEL, ACT_FM, stream);
        gemm(XEh, W16[p + 4], IN(p + 5), Vb, nullptr, M, D_MODEL, D_MODEL, ACT_NONE, stream);
        noncausal(Qb, Kb, Vb, Kb, ATTh);
        gemm(ATTh, W16[p + 6], IN(p + 7), Vb, nullptr, M, D_MODEL, D_MODEL, ACT_NONE, stream);
        ln_residual_kernel<<<lnGrid, b256, 0, stream>>>(XE, Vb, IN(p + 8), IN(p + 9), XE, XEh, M);
        gemm(XEh, W16[p + 12], IN(p + 13), H1, H1h, M, D_MODEL, FF_DIM, ACT_RELU, stream);
        gemm(H1h, W16[p + 14], IN(p + 15), Qb, nullptr, M, FF_DIM, D_MODEL, ACT_NONE, stream);
        ln_residual_kernel<<<lnGrid, b256, 0, stream>>>(XE, Qb, IN(p + 10), IN(p + 11), XE, XEh, M);
    }

    // ===================== Decoder =====================
    embed_kernel<<<(M * D_MODEL) / 256, b256, 0, stream>>>(tgt, teW, teB, XD, XDh, 3, M);

    for (int l = 0; l < 4; ++l) {
        const int p = 72 + 26 * l;
        // ---- causal self-attention (3-pass parallel chunked scan) ----
        gemm(XDh, W16[p + 0], IN(p + 1), Qb, nullptr, M, D_MODEL, D_MODEL, ACT_FM, stream);
        gemm(XDh, W16[p + 2], IN(p + 3), Kb, nullptr, M, D_MODEL, D_MODEL, ACT_FM, stream);
        gemm(XDh, W16[p + 4], IN(p + 5), Vb, nullptr, M, D_MODEL, D_MODEL, ACT_NONE, stream);
        causal_chunk_kv_kernel<<<dim3(NBH, NCHUNK), b256, 0, stream>>>(Kb, Vb, ChunkKV, ChunkKS, SEQ);
        causal_scan_kernel<<<scanGrid, b256, 0, stream>>>(ChunkKV, ChunkKS);
        causal_apply_kernel<<<dim3(NBH, NCHUNK), b256, 0, stream>>>(Qb, Kb, Vb, ChunkKV, ChunkKS, Qb, ATTh, SEQ);
        gemm(ATTh, W16[p + 6], IN(p + 7), Kb, nullptr, M, D_MODEL, D_MODEL, ACT_NONE, stream);
        ln_residual_kernel<<<lnGrid, b256, 0, stream>>>(XD, Kb, IN(p + 16), IN(p + 17), XD, XDh, M);
        // ---- cross attention (memory = XE) ----
        gemm(XDh, W16[p + 8],  IN(p + 9),  Qb, nullptr, M, D_MODEL, D_MODEL, ACT_FM, stream);
        gemm(XEh, W16[p + 10], IN(p + 11), Kb, nullptr, M, D_MODEL, D_MODEL, ACT_FM, stream);
        gemm(XEh, W16[p + 12], IN(p + 13), Vb, nullptr, M, D_MODEL, D_MODEL, ACT_NONE, stream);
        noncausal(Qb, Kb, Vb, Kb, ATTh);
        gemm(ATTh, W16[p + 14], IN(p + 15), Vb, nullptr, M, D_MODEL, D_MODEL, ACT_NONE, stream);
        ln_residual_kernel<<<lnGrid, b256, 0, stream>>>(XD, Vb, IN(p + 18), IN(p + 19), XD, XDh, M);
        // ---- FF block ----
        gemm(XDh, W16[p + 22], IN(p + 23), H1, H1h, M, D_MODEL, FF_DIM, ACT_RELU, stream);
        gemm(H1h, W16[p + 24], IN(p + 25), Qb, nullptr, M, FF_DIM, D_MODEL, ACT_NONE, stream);
        ln_residual_kernel<<<lnGrid, b256, 0, stream>>>(XD, Qb, IN(p + 20), IN(p + 21), XD, XDh, M);
    }

    // ===================== Final projection =====================
    final_kernel<<<(M * 3 + 255) / 256, b256, 0, stream>>>(XD, fW, fB, (float*)d_out, M);
}
